// TransformerMultiHeadAttention_910533067172
// MI455X (gfx1250) — compile-verified, matrix operand order checked by reading, not run on hardware
//
#include <hip/hip_runtime.h>
#include <hip/hip_bf16.h>

// ---------------------------------------------------------------------------
// MHA forward for MI455X (gfx1250, wave32, WMMA).
// B=2, S=2048, D=2048, H=16, hd=128.  All matmuls on v_wmma_f32_16x16x32_bf16.
// ---------------------------------------------------------------------------

#define S_ 2048
#define D_ 2048
#define H_ 16
#define HD_ 128
#define B_ 2
#define NEG_INF_ (-1e9f)

typedef __attribute__((ext_vector_type(16))) __bf16 v16bf;
typedef __attribute__((ext_vector_type(8)))  __bf16 v8bf;
typedef __attribute__((ext_vector_type(8)))  float  v8f;

__device__ __forceinline__ v8f wmma_bf16(v16bf a, v16bf b, v8f c) {
  // (neg_a, A, neg_b, B, c_mod, C, reuse_a, reuse_b)
  return __builtin_amdgcn_wmma_f32_16x16x32_bf16(false, a, false, b,
                                                 (short)0, c, false, false);
}

// Load a 16x32 bf16 fragment (A layout; B layout is symmetric with N<->M).
// Lane l holds row (l&15); k-chunks at kc=((l&16)?8:0) and kc+16.
// Works for both LDS (__shared__-derived) and global pointers.
__device__ __forceinline__ v16bf load_frag(const __bf16* base, int stride, int lane) {
  const int r  = lane & 15;
  const int kc = (lane & 16) ? 8 : 0;
  const __bf16* p = base + (size_t)r * stride + kc;
  v8bf lo = *(const v8bf*)p;
  v8bf hi = *(const v8bf*)(p + 16);
  return __builtin_shufflevector(lo, hi, 0,1,2,3,4,5,6,7,8,9,10,11,12,13,14,15);
}

// ---------------------------------------------------------------------------
// C[M,N] = A[M,K] * W[N,K]^T  (NT GEMM), fp32 in, fp32 accumulate via bf16 WMMA.
// MODE 0: out fp32 row-major [M,N]
// MODE 1: out bf16 head-major [B,H,S,hd]   (Q,K projections)
// MODE 2: out bf16 head-transposed [B,H,hd,S]  (V projection, for P*V frags)
// Block: 256 threads (8 waves), tile 128x128, BK=32.
// ---------------------------------------------------------------------------
template <int MODE>
__global__ __launch_bounds__(256)
void gemm_nt(const float* __restrict__ A, const float* __restrict__ W,
             void* __restrict__ out, int M, int N, int K) {
  __shared__ __bf16 lA[128 * 40];  // stride 40 (pad) keeps 16B-aligned chunks
  __shared__ __bf16 lB[128 * 40];

  const int tid  = threadIdx.x;
  const int lane = tid & 31;
  const int wave = tid >> 5;
  const int wm   = (wave & 1) * 64;   // wave sub-tile: 64 rows
  const int wn   = (wave >> 1) * 32;  // wave sub-tile: 32 cols
  const int bm   = blockIdx.y * 128;
  const int bn   = blockIdx.x * 128;

  v8f acc[4][2];
#pragma unroll
  for (int mt = 0; mt < 4; ++mt)
#pragma unroll
    for (int nt = 0; nt < 2; ++nt)
#pragma unroll
      for (int j = 0; j < 8; ++j) acc[mt][nt][j] = 0.0f;

  const int lr = tid >> 3;        // 0..31, row within pass
  const int lc = (tid & 7) * 4;   // k column (float4)

  for (int k0 = 0; k0 < K; k0 += 32) {
#pragma unroll
    for (int p = 0; p < 4; ++p) {
      const int r = lr + p * 32;
      const float4 a4 = *(const float4*)(A + (size_t)(bm + r) * K + k0 + lc);
      __bf16* da = lA + r * 40 + lc;
      da[0] = (__bf16)a4.x; da[1] = (__bf16)a4.y;
      da[2] = (__bf16)a4.z; da[3] = (__bf16)a4.w;
      const float4 b4 = *(const float4*)(W + (size_t)(bn + r) * K + k0 + lc);
      __bf16* db = lB + r * 40 + lc;
      db[0] = (__bf16)b4.x; db[1] = (__bf16)b4.y;
      db[2] = (__bf16)b4.z; db[3] = (__bf16)b4.w;
    }
    __syncthreads();

    v16bf af[4], bfr[2];
#pragma unroll
    for (int mt = 0; mt < 4; ++mt)
      af[mt] = load_frag(lA + (wm + mt * 16) * 40, 40, lane);
#pragma unroll
    for (int nt = 0; nt < 2; ++nt)
      bfr[nt] = load_frag(lB + (wn + nt * 16) * 40, 40, lane);
#pragma unroll
    for (int mt = 0; mt < 4; ++mt)
#pragma unroll
      for (int nt = 0; nt < 2; ++nt)
        acc[mt][nt] = wmma_bf16(af[mt], bfr[nt], acc[mt][nt]);
    __syncthreads();
  }

  // Epilogue. C layout: value v of acc tile -> row M = v + 8*(lane>>4), col N = lane&15.
  const int rh = (lane >> 4) * 8;
  const int cn = lane & 15;
#pragma unroll
  for (int mt = 0; mt < 4; ++mt)
#pragma unroll
    for (int nt = 0; nt < 2; ++nt)
#pragma unroll
      for (int v = 0; v < 8; ++v) {
        const int m = bm + wm + mt * 16 + v + rh;
        const int n = bn + wn + nt * 16 + cn;
        const float val = acc[mt][nt][v];
        if (MODE == 0) {
          ((float*)out)[(size_t)m * N + n] = val;
        } else if (MODE == 1) {
          const int bI = m >> 11, s = m & (S_ - 1);
          const int hI = n >> 7,  d = n & (HD_ - 1);
          ((__bf16*)out)[((((size_t)bI * H_ + hI) * S_) + s) * HD_ + d] = (__bf16)val;
        } else {
          const int bI = m >> 11, s = m & (S_ - 1);
          const int hI = n >> 7,  d = n & (HD_ - 1);
          ((__bf16*)out)[((((size_t)bI * H_ + hI) * HD_) + d) * S_ + s] = (__bf16)val;
        }
      }
}

// ---------------------------------------------------------------------------
// Flash attention: grid (S/128, B*H), 256 threads. Each wave owns 16 query
// rows; streams key tiles of 32 with online softmax. Q,K bf16 [B,H,S,hd];
// V bf16 transposed [B,H,hd,S]; output fp32 merged-head [B,S,D].
// ---------------------------------------------------------------------------
__global__ __launch_bounds__(256)
void flash_attn(const __bf16* __restrict__ Q, const __bf16* __restrict__ Kh,
                const __bf16* __restrict__ Vt, const int* __restrict__ pad,
                float* __restrict__ AO) {
  __shared__ __bf16 pT[8][16 * 40];  // per-wave P staging (16x32, stride 40)

  const int bh   = blockIdx.y;
  const int b    = bh >> 4;     // /H_
  const int h    = bh & 15;
  const int tid  = threadIdx.x;
  const int lane = tid & 31;
  const int wave = tid >> 5;
  const int q0   = blockIdx.x * 128 + wave * 16;   // wave's first query row
  const int col  = lane & 15;
  const int halfq = lane >> 4;                     // row-half of C layout

  const __bf16* Qb = Q  + (size_t)bh * S_ * HD_;
  const __bf16* Kb = Kh + (size_t)bh * S_ * HD_;
  const __bf16* Vb = Vt + (size_t)bh * HD_ * S_;
  __bf16* myP = pT[wave];

  // Preload this wave's Q fragments (16 rows x 128 d -> 4 frags).
  v16bf qf[4];
#pragma unroll
  for (int f = 0; f < 4; ++f)
    qf[f] = load_frag(Qb + (size_t)q0 * HD_ + f * 32, HD_, lane);

  v8f ao[8];
  float rowm[8], rowl[8];
#pragma unroll
  for (int i = 0; i < 8; ++i) {
    rowm[i] = -1e30f; rowl[i] = 0.0f;
#pragma unroll
    for (int j = 0; j < 8; ++j) ao[i][j] = 0.0f;
  }

  const float SCALE = 0.0883883476483184f;  // 1/sqrt(128)
  const int ntiles = q0 / 32 + 1;           // causal bound, wave-uniform

  for (int kt = 0; kt < ntiles; ++kt) {
    const int kbase = kt * 32;
    // Hint next K tile toward the caches while we compute this one.
    __builtin_prefetch(Kb + (size_t)(kbase + 32) * HD_ + lane * 8, 0, 0);

    // Scores: two 16x16 tiles over the 32-key strip (K-dim = hd = 128).
    v8f sc[2];
#pragma unroll
    for (int t = 0; t < 2; ++t) {
#pragma unroll
      for (int j = 0; j < 8; ++j) sc[t][j] = 0.0f;
#pragma unroll
      for (int f = 0; f < 4; ++f) {
        v16bf kf = load_frag(Kb + (size_t)(kbase + t * 16) * HD_ + f * 32, HD_, lane);
        sc[t] = wmma_bf16(qf[f], kf, sc[t]);
      }
    }

    const int key0 = kbase + col;
    const int key1 = kbase + 16 + col;
    const bool pk0 = pad[(size_t)b * S_ + key0] != 0;
    const bool pk1 = pad[(size_t)b * S_ + key1] != 0;

#pragma unroll
    for (int v = 0; v < 8; ++v) {
      const int row = q0 + v + 8 * halfq;
      float s0 = sc[0][v] * SCALE; if (key0 > row || !pk0) s0 = NEG_INF_;
      float s1 = sc[1][v] * SCALE; if (key1 > row || !pk1) s1 = NEG_INF_;
      float t = fmaxf(s0, s1);
#pragma unroll
      for (int m = 1; m < 16; m <<= 1) t = fmaxf(t, __shfl_xor(t, m, 32));
      const float newm = fmaxf(rowm[v], t);
      const float fac  = __expf(rowm[v] - newm);
      rowm[v] = newm;
      const float p0 = __expf(s0 - newm);
      const float p1 = __expf(s1 - newm);
      float ps = p0 + p1;
#pragma unroll
      for (int m = 1; m < 16; m <<= 1) ps += __shfl_xor(ps, m, 32);
      rowl[v] = rowl[v] * fac + ps;
#pragma unroll
      for (int nt = 0; nt < 8; ++nt) ao[nt][v] *= fac;
      myP[(v + 8 * halfq) * 40 + col]      = (__bf16)p0;
      myP[(v + 8 * halfq) * 40 + 16 + col] = (__bf16)p1;
    }

    // Wave-private LDS: make the 16x32 P tile visible before fragment reload.
    asm volatile("s_wait_dscnt 0" ::: "memory");

    v16bf pf = load_frag(myP, 40, lane);   // P as A-fragment (K = 32 keys)
#pragma unroll
    for (int nt = 0; nt < 8; ++nt) {
      v16bf vf = load_frag(Vb + (size_t)(nt * 16) * S_ + kbase, S_, lane);
      ao[nt] = wmma_bf16(pf, vf, ao[nt]);
    }
  }

  // Normalize and write merged-head output [B,S,D] fp32.
#pragma unroll
  for (int v = 0; v < 8; ++v) {
    const int row = q0 + v + 8 * halfq;
    const float inv = 1.0f / rowl[v];
    const size_t base = ((size_t)b * S_ + row) * D_ + h * HD_;
#pragma unroll
    for (int nt = 0; nt < 8; ++nt)
      AO[base + nt * 16 + col] = ao[nt][v] * inv;
  }
}

// ---------------------------------------------------------------------------
// Launch: 3 projection GEMMs -> flash attention -> output GEMM.
// Workspace: Qh(16MiB) Kh(16MiB) Vt(16MiB) bf16, AO(32MiB) fp32 = 80MiB.
// ---------------------------------------------------------------------------
extern "C" void kernel_launch(void* const* d_in, const int* in_sizes, int n_in,
                              void* d_out, int out_size, void* d_ws, size_t ws_size,
                              hipStream_t stream) {
  const float* q   = (const float*)d_in[0];
  const float* k   = (const float*)d_in[1];
  const float* v   = (const float*)d_in[2];
  const int*   pm  = (const int*)d_in[3];
  // d_in[4] = causal mask (analytic tril applied in-kernel)
  const float* Wq  = (const float*)d_in[5];
  const float* Wk  = (const float*)d_in[6];
  const float* Wv  = (const float*)d_in[7];
  const float* Wo  = (const float*)d_in[8];
  float* out = (float*)d_out;

  char* ws = (char*)d_ws;
  __bf16* Qh = (__bf16*)(ws);
  __bf16* Kh = (__bf16*)(ws + (size_t)16 * 1024 * 1024);
  __bf16* Vt = (__bf16*)(ws + (size_t)32 * 1024 * 1024);
  float*  AO = (float*) (ws + (size_t)48 * 1024 * 1024);

  const int M = B_ * S_;   // 4096
  const int N = D_;        // 2048
  const int K = D_;        // 2048
  dim3 gGemm(N / 128, M / 128);
  dim3 blk(256);

  gemm_nt<1><<<gGemm, blk, 0, stream>>>(q, Wq, (void*)Qh, M, N, K);
  gemm_nt<1><<<gGemm, blk, 0, stream>>>(k, Wk, (void*)Kh, M, N, K);
  gemm_nt<2><<<gGemm, blk, 0, stream>>>(v, Wv, (void*)Vt, M, N, K);

  dim3 gAttn(S_ / 128, B_ * H_);
  flash_attn<<<gAttn, blk, 0, stream>>>(Qh, Kh, Vt, pm, AO);

  gemm_nt<0><<<gGemm, blk, 0, stream>>>(AO, Wo, (void*)out, M, N, K);
}
